// Attention_27659589386447
// MI455X (gfx1250) — matrix operand order repaired and checked by reading, not compiled, not run on hardware
//
#include <hip/hip_runtime.h>

typedef __attribute__((ext_vector_type(16))) _Float16 v16h;
typedef __attribute__((ext_vector_type(8)))  float    v8f;

#define NB 2
#define NS 2048
#define ND 1024
#define NH 16
#define NHD 64

static __device__ __forceinline__ v8f wmma_f16(v16h a, v16h b, v8f c) {
  // D(16x16 f32) = A(16x32 f16) x B(32x16 f16) + C
  return __builtin_amdgcn_wmma_f32_16x16x32_f16(false, a, false, b, (short)0, c,
                                                false, false);
}

// ---------------------------------------------------------------------------
// Kernel 1: projection (x @ W^T) + optional RoPE, output f16 [B,H,S,HD].
// One wave computes a 32(seq) x 64(one head) tile: two A-tiles reuse every
// B-tile (8 WMMAs per 32-deep K-step). blockIdx.z = b*2 + mat.
// ---------------------------------------------------------------------------
__global__ __launch_bounds__(128) void proj_rope_kernel(
    const float* __restrict__ x, const float* __restrict__ Wq,
    const float* __restrict__ Wv, const float* __restrict__ cosT,
    const float* __restrict__ sinT, _Float16* __restrict__ Qbuf,
    _Float16* __restrict__ Vbuf) {
  const int lane = threadIdx.x & 31;
  const int wave = threadIdx.x >> 5;
  const int hi = lane >> 4;   // 0: lanes 0-15, 1: lanes 16-31
  const int ln = lane & 15;
  const int s0 = (blockIdx.x * 4 + wave) * 32;
  const int h  = blockIdx.y;
  const int b  = blockIdx.z >> 1;
  const int mat = blockIdx.z & 1;
  const float* W = mat ? Wv : Wq;
  _Float16* outp = mat ? Vbuf : Qbuf;
  const int o0 = h * NHD;

  v8f acc[2][4] = {};  // two 16x64 f32 accumulators (s0.., s0+16..)

  const float* xrow0 = x + ((size_t)b * NS + s0 + ln) * ND;
  const float* xrow1 = xrow0 + (size_t)16 * ND;

  for (int kb = 0; kb < ND / 32; ++kb) {
    const int d0 = kb * 32;
    v16h a0, a1;
#pragma unroll
    for (int j = 0; j < 16; ++j) {
      const int K = hi * 16 + j;
      a0[j] = (_Float16)xrow0[d0 + K];
      a1[j] = (_Float16)xrow1[d0 + K];
    }
    if (kb + 1 < ND / 32) {  // uniform branch: EXEC stays all-ones
      __builtin_prefetch(xrow0 + d0 + 32, 0, 0);
      __builtin_prefetch(xrow1 + d0 + 32, 0, 0);
    }
#pragma unroll
    for (int c = 0; c < 4; ++c) {
      // B-layout (32x16, k=d, n=o): lane holds col o0+16c+ln, k = hi*16 + t.
      const float* wrow = W + (size_t)(o0 + 16 * c + ln) * ND + d0 + hi * 16;
      v16h bm;
#pragma unroll
      for (int t = 0; t < 16; ++t) bm[t] = (_Float16)wrow[t];
      if (kb + 1 < ND / 32) __builtin_prefetch(wrow + 32, 0, 0);
      acc[0][c] = wmma_f16(a0, bm, acc[0][c]);
      acc[1][c] = wmma_f16(a1, bm, acc[1][c]);
    }
  }

#pragma unroll
  for (int t = 0; t < 2; ++t) {
    if (mat == 0) {  // RoPE: pair d and d+32 -> C-tile chunks (c, c+2)
#pragma unroll
      for (int g = 0; g < 8; ++g) {
        const int pos = s0 + 16 * t + g + hi * 8;  // C layout: M = g + 8*hi
#pragma unroll
        for (int c = 0; c < 2; ++c) {
          const int dlo = 16 * c + ln;
          const float cl = cosT[pos * NHD + dlo];
          const float sl = sinT[pos * NHD + dlo];
          const float ch = cosT[pos * NHD + dlo + 32];
          const float sh = sinT[pos * NHD + dlo + 32];
          const float lo = acc[t][c][g], up = acc[t][c + 2][g];
          acc[t][c][g]     = lo * cl - up * sl;
          acc[t][c + 2][g] = up * ch + lo * sh;
        }
      }
    }
    _Float16* orow = outp + ((size_t)(b * NH + h) * NS) * NHD;
#pragma unroll
    for (int c = 0; c < 4; ++c)
#pragma unroll
      for (int g = 0; g < 8; ++g)
        orow[(size_t)(s0 + 16 * t + g + hi * 8) * NHD + 16 * c + ln] =
            (_Float16)acc[t][c][g];
  }
}

// ---------------------------------------------------------------------------
// Kernel 2: causal flash attention. K == Q (reference uses Wq for both).
// One wave per 32-query tile; 32 keys per inner step. K/V B-tiles are loaded
// once and reused by both 16-query sub-tiles (16 WMMAs / step).
// ---------------------------------------------------------------------------
__global__ __launch_bounds__(128) void flash_attn_kernel(
    const _Float16* __restrict__ Qbuf, const _Float16* __restrict__ Vbuf,
    float* __restrict__ out) {
  __shared__ _Float16 pbuf[4][2 * 16 * 32];  // per-wave P bounce (C->A relayout)
  const int lane = threadIdx.x & 31;
  const int wave = threadIdx.x >> 5;
  const int hi = lane >> 4;
  const int ln = lane & 15;
  const int q0 = (blockIdx.x * 4 + wave) * 32;
  const int h = blockIdx.y;
  const int b = blockIdx.z;
  const _Float16* Qh = Qbuf + ((size_t)(b * NH + h) * NS) * NHD;
  const _Float16* Vh = Vbuf + ((size_t)(b * NH + h) * NS) * NHD;
  _Float16* pb = pbuf[wave];

  // Q tiles as A matrices: A[qtile][dchunk], resident for the whole loop.
  v16h A[2][2];
#pragma unroll
  for (int t = 0; t < 2; ++t) {
    const _Float16* qrow = Qh + (size_t)(q0 + 16 * t + ln) * NHD;
#pragma unroll
    for (int j = 0; j < 16; ++j) {
      const int K = hi * 16 + j;
      A[t][0][j] = qrow[K];
      A[t][1][j] = qrow[K + 32];
    }
  }

  v8f acc[2][4] = {};
  float mrow[2][8], lrow[2][8];
#pragma unroll
  for (int t = 0; t < 2; ++t)
#pragma unroll
    for (int g = 0; g < 8; ++g) { mrow[t][g] = -1e30f; lrow[t][g] = 0.f; }

  const float scale = 0.125f;     // 1/sqrt(64)
  const int nkt = q0 / 32 + 1;    // key tiles covering 0..q0+31

  for (int kt = 0; kt < nkt; ++kt) {
    const int k0 = kt * 32;
    if (kt + 1 < nkt) {  // uniform: prefetch next K/V block (one row per lane)
      __builtin_prefetch(Qh + (size_t)(k0 + 32 + lane) * NHD, 0, 0);
      __builtin_prefetch(Vh + (size_t)(k0 + 32 + lane) * NHD, 0, 0);
    }
    // ---- K-block B tiles, loaded once, shared by both query tiles --------
    v16h KB[2][2];  // [16-key subtile][d chunk]
#pragma unroll
    for (int j = 0; j < 2; ++j) {
      const _Float16* krow = Qh + (size_t)(k0 + 16 * j + ln) * NHD + hi * 16;
#pragma unroll
      for (int u = 0; u < 16; ++u) {
        KB[j][0][u] = krow[u];
        KB[j][1][u] = krow[u + 32];
      }
    }
    // ---- scores ----------------------------------------------------------
    v8f st[2][2];
#pragma unroll
    for (int t = 0; t < 2; ++t)
#pragma unroll
      for (int j = 0; j < 2; ++j) {
        v8f cz = {};
        cz = wmma_f16(A[t][0], KB[j][0], cz);
        cz = wmma_f16(A[t][1], KB[j][1], cz);
        st[t][j] = cz;
      }
    // ---- causal mask + scale + online softmax ----------------------------
#pragma unroll
    for (int t = 0; t < 2; ++t) {
#pragma unroll
      for (int j = 0; j < 2; ++j)
#pragma unroll
        for (int g = 0; g < 8; ++g) {
          const int key = k0 + 16 * j + ln;            // C layout: N = ln
          const int q   = q0 + 16 * t + g + hi * 8;    //           M = g + 8*hi
          const float v = st[t][j][g] * scale;
          st[t][j][g] = (key > q) ? -1e30f : v;
        }
#pragma unroll
      for (int g = 0; g < 8; ++g) {
        float rm = fmaxf(st[t][0][g], st[t][1][g]);
        rm = fmaxf(rm, __shfl_xor(rm, 1, 32));
        rm = fmaxf(rm, __shfl_xor(rm, 2, 32));
        rm = fmaxf(rm, __shfl_xor(rm, 4, 32));
        rm = fmaxf(rm, __shfl_xor(rm, 8, 32));
        const float mnew  = fmaxf(mrow[t][g], rm);
        const float alpha = __expf(mrow[t][g] - mnew);
        mrow[t][g] = mnew;
        const float p0 = __expf(st[t][0][g] - mnew);
        const float p1 = __expf(st[t][1][g] - mnew);
        st[t][0][g] = p0; st[t][1][g] = p1;
        float rs = p0 + p1;
        rs += __shfl_xor(rs, 1, 32);
        rs += __shfl_xor(rs, 2, 32);
        rs += __shfl_xor(rs, 4, 32);
        rs += __shfl_xor(rs, 8, 32);
        lrow[t][g] = lrow[t][g] * alpha + rs;
#pragma unroll
        for (int c = 0; c < 4; ++c) acc[t][c][g] *= alpha;
      }
    }
    // ---- P: C-layout -> LDS -> A-layout ----------------------------------
#pragma unroll
    for (int t = 0; t < 2; ++t)
#pragma unroll
      for (int j = 0; j < 2; ++j)
#pragma unroll
        for (int g = 0; g < 8; ++g)
          pb[t * 512 + (g + hi * 8) * 32 + 16 * j + ln] =
              (_Float16)st[t][j][g];
    asm volatile("s_wait_dscnt 0" ::: "memory");  // cross-lane LDS RAW
    v16h Pa[2];
#pragma unroll
    for (int t = 0; t < 2; ++t)
#pragma unroll
      for (int j = 0; j < 16; ++j) {
        const int K = hi * 16 + j;
        Pa[t][j] = pb[t * 512 + ln * 32 + K];
      }
    // ---- P @ V: V B-tiles loaded once, shared by both query tiles --------
#pragma unroll
    for (int c = 0; c < 4; ++c) {
      v16h Bv;  // B(k=key, n=d): lane holds d = 16c+ln, key = k0 + hi*16 + u
#pragma unroll
      for (int u = 0; u < 16; ++u)
        Bv[u] = Vh[(size_t)(k0 + hi * 16 + u) * NHD + 16 * c + ln];
      acc[0][c] = wmma_f16(Pa[0], Bv, acc[0][c]);
      acc[1][c] = wmma_f16(Pa[1], Bv, acc[1][c]);
    }
    asm volatile("s_wait_dscnt 0" ::: "memory");  // Pa reads before next store
  }

  // ---- epilogue: normalize and write fp32 [B,S, h*HD + d] ----------------
#pragma unroll
  for (int t = 0; t < 2; ++t)
#pragma unroll
    for (int g = 0; g < 8; ++g) {
      const float inv = 1.0f / lrow[t][g];
      const int s = q0 + 16 * t + g + hi * 8;
#pragma unroll
      for (int c = 0; c < 4; ++c)
        __builtin_nontemporal_store(
            acc[t][c][g] * inv,
            out + (size_t)(b * NS + s) * ND + h * NHD + 16 * c + ln);
    }
}

// ---------------------------------------------------------------------------
extern "C" void kernel_launch(void* const* d_in, const int* in_sizes, int n_in,
                              void* d_out, int out_size, void* d_ws,
                              size_t ws_size, hipStream_t stream) {
  const float* x    = (const float*)d_in[0];
  const float* Wq   = (const float*)d_in[1];
  const float* Wv   = (const float*)d_in[2];
  const float* cosT = (const float*)d_in[3];
  const float* sinT = (const float*)d_in[4];
  float* out = (float*)d_out;

  _Float16* Qbuf = (_Float16*)d_ws;                       // 8 MB
  _Float16* Vbuf = Qbuf + (size_t)NB * NH * NS * NHD;     // 8 MB

  // Projection: 32-row s-tiles, 4 waves/block -> grid (S/128, H, B*2 {Q,V})
  proj_rope_kernel<<<dim3(NS / 128, NH, NB * 2), 128, 0, stream>>>(
      x, Wq, Wv, cosT, sinT, Qbuf, Vbuf);
  // Attention: 32-query tiles, 4 waves/block -> grid (S/128, H, B)
  flash_attn_kernel<<<dim3(NS / 128, NH, NB), 128, 0, stream>>>(Qbuf, Vbuf,
                                                                out);
}
